// HyperDualAttention_56152402428200
// MI455X (gfx1250) — matrix operand order repaired and checked by reading, not compiled, not run on hardware
//
#include <hip/hip_runtime.h>
#include <hip/hip_bf16.h>

// ---------------------------------------------------------------------------
// HyperDualAttention for MI455X (gfx1250, wave32, WMMA bf16)
//
//   P0/P1 prep_pack      : conv weights -> per-tap bf16 [tap][co][ci]; zero gsum
//   K1    conv3x3_wmma<f32-in, PReLU, no-stats> : x -> r1 (bf16)
//   K2    conv3x3_wmma<bf16-in, no-PReLU,stats> : r1 -> r (bf16) + avg/max maps
//                                                 + global channel sums
//   K3    tiny_mlp       : channel-attention MLP + hyper net -> Kp bf16 [B][64][128]
//   K4    sa_conv        : 2->1 3x3 reflect conv + sigmoid
//   K5    hyper_out_wmma : rc=[r*sa ; r*ca] (LDS, K-contiguous), GEMM with Kp,
//                          + hc_bias + residual x -> out (f32)
//
// GEMM blocking: 128 threads = 4 waves, wave tile 32x64 (2 M-subtiles x
// 4 N-subtiles, 8 f32 accumulators). Per K-step: 2 A-frag pairs
// (global_load_b128, L2-resident weights) + 4 B-frag pairs (ds_load_b128,
// K-contiguous LDS, conflict-free pitch) feed 8 back-to-back WMMAs
// (1.5 x b128 loads per WMMA; every fragment reused).
// ---------------------------------------------------------------------------

typedef __attribute__((ext_vector_type(16))) __bf16          v16bf;
typedef __attribute__((ext_vector_type(8)))  float           v8f;
typedef __attribute__((ext_vector_type(8)))  unsigned short  u16x8;
typedef __attribute__((ext_vector_type(4)))  float           f32x4;

#define HH 256
#define WW 256
#define BB 8
#define CC 64
#define THREADS 128
#define INPITCH 72   // ushorts per (row,col) channel vector: 144B -> bank-rotating
#define RCPITCH 136  // ushorts per pixel in rc tile: 272B -> +4 banks per pixel

union FragU { v16bf v; u16x8 h[2]; };

__device__ __forceinline__ unsigned short f2bf(float f) {
  unsigned int u = __builtin_bit_cast(unsigned int, f);
  unsigned int r = u + 0x7FFFu + ((u >> 16) & 1u);  // round-to-nearest-even
  return (unsigned short)(r >> 16);
}
__device__ __forceinline__ float bf2f(unsigned short u) {
  return __builtin_bit_cast(float, (unsigned int)u << 16);
}
__device__ __forceinline__ int refl(int i, int n) {
  if (i < 0) i = -i;
  if (i >= n) i = 2 * n - 2 - i;
  return i;
}

// ---------------------------------------------------------------------------
// P0/P1: repack conv weights OIHW f32 -> [tap][co][ci] bf16; zero gsum
// ---------------------------------------------------------------------------
__global__ __launch_bounds__(256) void prep_pack(const float* __restrict__ w,
                                                 unsigned short* __restrict__ Wp,
                                                 int n, float* gsum, int ng) {
  int i = blockIdx.x * 256 + threadIdx.x;
  if (i < n) {
    int tap = i >> 12;          // kh*3+kw
    int co = (i >> 6) & 63;
    int ci = i & 63;
    Wp[i] = f2bf(w[co * 576 + ci * 9 + tap]);
  }
  if (gsum && i < ng) gsum[i] = 0.f;
}

// ---------------------------------------------------------------------------
// K1/K2: 3x3 reflect conv = 9 shifted 1x1 GEMMs (K=64 each), WMMA bf16.
//   Block tile: M=64 out-ch x N=128 pixels. 4 waves (2M x 2N), 32x64 each.
// ---------------------------------------------------------------------------
template <bool IN_BF16, bool PRELU, bool STATS>
__global__ __launch_bounds__(THREADS) void conv3x3_wmma(
    const void* __restrict__ in_, const unsigned short* __restrict__ Wp,
    const float* __restrict__ bias, const float* __restrict__ pa,
    unsigned short* __restrict__ outbf,
    float* __restrict__ avg_out, float* __restrict__ mx_out,
    float* __restrict__ gsum) {
  __shared__ union {
    unsigned short in[3 * 132 * INPITCH];  // [rr][cc][ci] channel-contiguous
    float rtile[CC * 128];                 // f32 result tile (post-K reuse)
  } smem;

  const int tid = threadIdx.x;
  const int lane = tid & 31;
  const int wid = tid >> 5;
  const int waveM = wid & 1;   // 32-channel strip
  const int waveN = wid >> 1;  // 64-pixel half
  const int b = blockIdx.z;
  const int h = blockIdx.y;
  const int w0 = blockIdx.x * 128;

  // ---- stage halo tile (reflect pad), transposed to channel-contiguous ---
  for (int e = tid; e < CC * 3 * 130; e += THREADS) {
    int ci = e / 390;
    int rem = e - ci * 390;
    int rr = rem / 130;
    int cc = rem - rr * 130;
    int gh = refl(h - 1 + rr, HH);
    int gw = refl(w0 - 1 + cc, WW);
    long gidx = (((long)b * CC + ci) * HH + gh) * WW + gw;
    unsigned short bits;
    if constexpr (IN_BF16)
      bits = ((const unsigned short*)in_)[gidx];
    else
      bits = f2bf(((const float*)in_)[gidx]);
    smem.in[(rr * 132 + cc) * INPITCH + ci] = bits;
  }
  __syncthreads();

  v8f acc[2][4] = {};

  const int rowA = waveM * 32 + (lane & 15);  // M-subtile 0 row; +16 for subtile 1
  const int koff = (lane < 16) ? 0 : 8;       // A-frag K sub-offset
  const int kboff = (lane < 16) ? 0 : 8;      // B-frag K sub-offset
  const int nlane = lane & 15;

  // ---- 9 taps x 2 K-steps of 32 over input channels ----------------------
  for (int tap = 0; tap < 9; ++tap) {
    const int kh = tap / 3;
    const int kw = tap - kh * 3;
    const unsigned short* wtap = Wp + tap * 4096;
#pragma unroll
    for (int k0 = 0; k0 < 64; k0 += 32) {
      FragU A0, A1, Bf[4];
      {
        const unsigned short* wr0 = wtap + rowA * 64 + k0;
        A0.h[0] = *(const u16x8*)(wr0 + koff);
        A0.h[1] = *(const u16x8*)(wr0 + 16 + koff);
        const unsigned short* wr1 = wr0 + 16 * 64;
        A1.h[0] = *(const u16x8*)(wr1 + koff);
        A1.h[1] = *(const u16x8*)(wr1 + 16 + koff);
      }
#pragma unroll
      for (int t = 0; t < 4; ++t) {
        const int px = waveN * 64 + t * 16 + nlane;
        const unsigned short* bp =
            &smem.in[(kh * 132 + px + kw) * INPITCH + k0 + kboff];
        Bf[t].h[0] = *(const u16x8*)bp;
        Bf[t].h[1] = *(const u16x8*)(bp + 16);
      }
#pragma unroll
      for (int t = 0; t < 4; ++t)
        acc[0][t] = __builtin_amdgcn_wmma_f32_16x16x32_bf16(
            false, A0.v, false, Bf[t].v, (short)0, acc[0][t], false, false);
#pragma unroll
      for (int t = 0; t < 4; ++t)
        acc[1][t] = __builtin_amdgcn_wmma_f32_16x16x32_bf16(
            false, A1.v, false, Bf[t].v, (short)0, acc[1][t], false, false);
    }
  }

  __syncthreads();  // all LDS B reads done; reuse union as f32 result tile

  // ---- epilogue: bias (+PReLU) -> f32 LDS tile ---------------------------
  const int mofs = (lane < 16) ? 0 : 8;
#pragma unroll
  for (int i = 0; i < 2; ++i) {
#pragma unroll
    for (int t = 0; t < 4; ++t) {
      const int px = waveN * 64 + t * 16 + nlane;
#pragma unroll
      for (int r = 0; r < 8; ++r) {
        const int co = waveM * 32 + i * 16 + mofs + r;
        float v = acc[i][t][r] + bias[co];
        if constexpr (PRELU) v = fmaxf(v, 0.f) + pa[co] * fminf(v, 0.f);
        smem.rtile[co * 128 + px] = v;
      }
    }
  }
  __syncthreads();

  // ---- coalesced bf16 stores: 128-bit per lane ---------------------------
  for (int e = tid; e < CC * 128 / 8; e += THREADS) {
    int row = e >> 4;
    int seg = e & 15;
    u16x8 pk;
#pragma unroll
    for (int j = 0; j < 8; ++j) pk[j] = f2bf(smem.rtile[row * 128 + seg * 8 + j]);
    long base = (((long)b * CC + row) * HH + h) * WW + (w0 + seg * 8);
    *(u16x8*)(outbf + base) = pk;
  }

  if constexpr (STATS) {
    // per-pixel channel mean / max (spatial attention inputs)
    {
      float s = 0.f, m = -3.4e38f;
#pragma unroll 8
      for (int ci = 0; ci < CC; ++ci) {
        float v = smem.rtile[ci * 128 + tid];
        s += v;
        m = fmaxf(m, v);
      }
      long p = ((long)b * HH + h) * WW + (w0 + tid);
      avg_out[p] = s * (1.f / CC);
      mx_out[p] = m;
    }
    // per-channel partial sums (channel attention global pool)
    if (tid < CC) {
      float s = 0.f;
#pragma unroll 8
      for (int px = 0; px < 128; ++px) s += smem.rtile[tid * 128 + px];
      atomicAdd(&gsum[b * CC + tid], s);
    }
  }
}

// ---------------------------------------------------------------------------
// K3: channel attention MLP + hyper kernel net (one block per batch sample)
// ---------------------------------------------------------------------------
__global__ __launch_bounds__(64) void tiny_mlp(
    const float* __restrict__ gsum, const float* __restrict__ ca_w1,
    const float* __restrict__ ca_b1, const float* __restrict__ ca_a,
    const float* __restrict__ ca_w2, const float* __restrict__ ca_b2,
    const float* __restrict__ hvec, const float* __restrict__ fc_w,
    const float* __restrict__ fc_b, const float* __restrict__ k1_w,
    const float* __restrict__ k1_b, const float* __restrict__ k2_w,
    const float* __restrict__ k2_b, const float* __restrict__ k3_w,
    const float* __restrict__ k3_b, float* __restrict__ catt,
    unsigned short* __restrict__ Kp) {
  const int b = blockIdx.x, t = threadIdx.x;
  __shared__ float g[64], s1[32], t0[16], t1[32], t2[32];
  g[t] = gsum[b * 64 + t] * (1.f / (HH * WW));
  __syncthreads();
  if (t < 32) {
    float a = ca_b1[t];
    for (int i = 0; i < 64; ++i) a += ca_w1[t * 64 + i] * g[i];
    s1[t] = fmaxf(a, 0.f) + ca_a[t] * fminf(a, 0.f);
  }
  if (t < 16) {
    float a = fc_b[t];
    for (int i = 0; i < 16; ++i) a += fc_w[t * 16 + i] * hvec[b * 16 + i];
    t0[t] = a >= 0.f ? a : 0.01f * a;
  }
  __syncthreads();
  {
    float a = ca_b2[t];
    for (int i = 0; i < 32; ++i) a += ca_w2[t * 32 + i] * s1[i];
    catt[b * 64 + t] = 1.f / (1.f + __expf(-a));
  }
  if (t < 32) {
    float a = k1_b[t];
    for (int i = 0; i < 16; ++i) a += k1_w[t * 16 + i] * t0[i];
    t1[t] = a >= 0.f ? a : 0.01f * a;
  }
  __syncthreads();
  if (t < 32) {
    float a = k2_b[t];
    for (int i = 0; i < 32; ++i) a += k2_w[t * 32 + i] * t1[i];
    t2[t] = a >= 0.f ? a : 0.01f * a;
  }
  __syncthreads();
  // Kp[b][o=t][i=j]: per-sample A-matrix for final GEMM, pre-packed bf16
  for (int j = 0; j < 128; ++j) {
    int idx = t * 128 + j;
    float a = k3_b[idx];
    for (int i = 0; i < 32; ++i) a += k3_w[idx * 32 + i] * t2[i];
    Kp[b * 8192 + idx] = f2bf(a);
  }
}

// ---------------------------------------------------------------------------
// K4: spatial attention conv (2->1 ch, 3x3 reflect) + sigmoid
// ---------------------------------------------------------------------------
__global__ __launch_bounds__(256) void sa_conv(const float* __restrict__ avg,
                                               const float* __restrict__ mx,
                                               const float* __restrict__ sa_w,
                                               const float* __restrict__ sa_b,
                                               float* __restrict__ ssig) {
  int p = blockIdx.x * 256 + threadIdx.x;
  if (p >= BB * HH * WW) return;
  int b = p >> 16, hh = (p >> 8) & 255, ww = p & 255;
  const float* ba = avg + ((long)b << 16);
  const float* bm = mx + ((long)b << 16);
  float s = sa_b[0];
#pragma unroll
  for (int dh = 0; dh < 3; ++dh) {
    int gh = refl(hh + dh - 1, HH);
#pragma unroll
    for (int dw = 0; dw < 3; ++dw) {
      int gw = refl(ww + dw - 1, WW);
      int o = (gh << 8) + gw;
      s += sa_w[dh * 3 + dw] * ba[o] + sa_w[9 + dh * 3 + dw] * bm[o];
    }
  }
  ssig[p] = 1.f / (1.f + __expf(-s));
}

// ---------------------------------------------------------------------------
// K5: fused gating + per-sample 1x1 hyper conv (WMMA, K=128) + bias + residual
// ---------------------------------------------------------------------------
__global__ __launch_bounds__(THREADS) void hyper_out_wmma(
    const unsigned short* __restrict__ rbf, const float* __restrict__ ssig,
    const float* __restrict__ catt, const unsigned short* __restrict__ Kp,
    const float* __restrict__ hc_bias, const float* __restrict__ x,
    float* __restrict__ out) {
  __shared__ union {
    unsigned short rc[128 * RCPITCH];  // [px][128 ch] K-contiguous bf16
    float otile[CC * 128];             // f32 output tile (post-K reuse)
  } smem;
  __shared__ float ssl[128];
  __shared__ float ctl[64];

  const int tid = threadIdx.x;
  const int lane = tid & 31;
  const int wid = tid >> 5;
  const int waveM = wid & 1;
  const int waveN = wid >> 1;
  const int b = blockIdx.z;
  const int h = blockIdx.y;
  const int w0 = blockIdx.x * 128;

  if (tid < 128) ssl[tid] = ssig[((long)b * HH + h) * WW + (w0 + tid)];
  if (tid < 64) ctl[tid] = catt[b * 64 + tid];
  __syncthreads();

  // build rc = concat([r*sigmoid(s), r*sigmoid(g)]) : K-contiguous per pixel
  for (int e = tid; e < CC * 128 / 8; e += THREADS) {
    int ci = e >> 4;
    int seg = e & 15;
    long base = (((long)b * CC + ci) * HH + h) * WW + (w0 + seg * 8);
    u16x8 rv8 = *(const u16x8*)(rbf + base);
#pragma unroll
    for (int j = 0; j < 8; ++j) {
      int px = seg * 8 + j;
      float rv = bf2f(rv8[j]);
      smem.rc[px * RCPITCH + ci] = f2bf(rv * ssl[px]);
      smem.rc[px * RCPITCH + CC + ci] = f2bf(rv * ctl[ci]);
    }
  }
  __syncthreads();

  v8f acc[2][4] = {};
  const unsigned short* Kb = Kp + b * 8192;
  const int rowA = waveM * 32 + (lane & 15);
  const int koff = (lane < 16) ? 0 : 8;
  const int kboff = (lane < 16) ? 0 : 8;
  const int nlane = lane & 15;

#pragma unroll
  for (int k0 = 0; k0 < 128; k0 += 32) {
    FragU A0, A1, Bf[4];
    {
      const unsigned short* wr0 = Kb + rowA * 128 + k0;
      A0.h[0] = *(const u16x8*)(wr0 + koff);
      A0.h[1] = *(const u16x8*)(wr0 + 16 + koff);
      const unsigned short* wr1 = wr0 + 16 * 128;
      A1.h[0] = *(const u16x8*)(wr1 + koff);
      A1.h[1] = *(const u16x8*)(wr1 + 16 + koff);
    }
#pragma unroll
    for (int t = 0; t < 4; ++t) {
      const int px = waveN * 64 + t * 16 + nlane;
      const unsigned short* bp = &smem.rc[px * RCPITCH + k0 + kboff];
      Bf[t].h[0] = *(const u16x8*)bp;
      Bf[t].h[1] = *(const u16x8*)(bp + 16);
    }
#pragma unroll
    for (int t = 0; t < 4; ++t)
      acc[0][t] = __builtin_amdgcn_wmma_f32_16x16x32_bf16(
          false, A0.v, false, Bf[t].v, (short)0, acc[0][t], false, false);
#pragma unroll
    for (int t = 0; t < 4; ++t)
      acc[1][t] = __builtin_amdgcn_wmma_f32_16x16x32_bf16(
          false, A1.v, false, Bf[t].v, (short)0, acc[1][t], false, false);
  }

  __syncthreads();  // rc reads done; reuse union as f32 output tile

  const int mofs = (lane < 16) ? 0 : 8;
#pragma unroll
  for (int i = 0; i < 2; ++i) {
#pragma unroll
    for (int t = 0; t < 4; ++t) {
      const int px = waveN * 64 + t * 16 + nlane;
#pragma unroll
      for (int r = 0; r < 8; ++r) {
        const int co = waveM * 32 + i * 16 + mofs + r;
        smem.otile[co * 128 + px] = acc[i][t][r];
      }
    }
  }
  __syncthreads();

  // coalesced f32x4: out = x + y + hc_bias
  for (int e = tid; e < CC * 128 / 4; e += THREADS) {
    int row = e >> 5;
    int seg = e & 31;
    long base = (((long)b * CC + row) * HH + h) * WW + (w0 + seg * 4);
    f32x4 xv = *(const f32x4*)(x + base);
    f32x4 yv = *(const f32x4*)&smem.otile[row * 128 + seg * 4];
    f32x4 o = xv + yv + hc_bias[row];
    *(f32x4*)(out + base) = o;
  }
}

// ---------------------------------------------------------------------------
extern "C" void kernel_launch(void* const* d_in, const int* in_sizes, int n_in,
                              void* d_out, int out_size, void* d_ws,
                              size_t ws_size, hipStream_t stream) {
  const float* x       = (const float*)d_in[0];
  const float* h       = (const float*)d_in[1];
  const float* conv1_w = (const float*)d_in[2];
  const float* conv1_b = (const float*)d_in[3];
  const float* prelu_a = (const float*)d_in[4];
  const float* conv2_w = (const float*)d_in[5];
  const float* conv2_b = (const float*)d_in[6];
  const float* sa_w    = (const float*)d_in[7];
  const float* sa_b    = (const float*)d_in[8];
  const float* ca_w1   = (const float*)d_in[9];
  const float* ca_b1   = (const float*)d_in[10];
  const float* ca_a    = (const float*)d_in[11];
  const float* ca_w2   = (const float*)d_in[12];
  const float* ca_b2   = (const float*)d_in[13];
  const float* fc_w    = (const float*)d_in[14];
  const float* fc_b    = (const float*)d_in[15];
  const float* k1_w    = (const float*)d_in[16];
  const float* k1_b    = (const float*)d_in[17];
  const float* k2_w    = (const float*)d_in[18];
  const float* k2_b    = (const float*)d_in[19];
  const float* k3_w    = (const float*)d_in[20];
  const float* k3_b    = (const float*)d_in[21];
  const float* hc_bias = (const float*)d_in[22];
  float* out = (float*)d_out;

  char* ws = (char*)d_ws;
  size_t off = 0;
  auto alloc = [&](size_t bytes) -> char* {
    char* p = ws + off;
    off = (off + bytes + 255) & ~(size_t)255;
    return p;
  };
  const size_t NMAP = (size_t)BB * HH * WW;  // 524288
  unsigned short* Wp1 = (unsigned short*)alloc(64 * 576 * 2);
  unsigned short* Wp2 = (unsigned short*)alloc(64 * 576 * 2);
  unsigned short* r1  = (unsigned short*)alloc((size_t)BB * CC * HH * WW * 2);
  unsigned short* r   = (unsigned short*)alloc((size_t)BB * CC * HH * WW * 2);
  float* avg  = (float*)alloc(NMAP * 4);
  float* mx   = (float*)alloc(NMAP * 4);
  float* ssig = (float*)alloc(NMAP * 4);
  float* gsum = (float*)alloc(BB * CC * 4);
  float* catt = (float*)alloc(BB * CC * 4);
  unsigned short* Kp = (unsigned short*)alloc((size_t)BB * 8192 * 2);

  const int NW = 64 * 576;  // 36864 weight elements per conv
  prep_pack<<<(NW + 255) / 256, 256, 0, stream>>>(conv1_w, Wp1, NW, gsum,
                                                  BB * CC);
  prep_pack<<<(NW + 255) / 256, 256, 0, stream>>>(conv2_w, Wp2, NW, nullptr, 0);

  dim3 grid(WW / 128, HH, BB);  // (2, 256, 8)
  conv3x3_wmma<false, true, false><<<grid, THREADS, 0, stream>>>(
      (const void*)x, Wp1, conv1_b, prelu_a, r1, nullptr, nullptr, nullptr);
  conv3x3_wmma<true, false, true><<<grid, THREADS, 0, stream>>>(
      (const void*)r1, Wp2, conv2_b, nullptr, r, avg, mx, gsum);

  tiny_mlp<<<BB, 64, 0, stream>>>(gsum, ca_w1, ca_b1, ca_a, ca_w2, ca_b2, h,
                                  fc_w, fc_b, k1_w, k1_b, k2_w, k2_b, k3_w,
                                  k3_b, catt, Kp);

  sa_conv<<<(int)(NMAP / 256), 256, 0, stream>>>(avg, mx, sa_w, sa_b, ssig);

  hyper_out_wmma<<<grid, THREADS, 0, stream>>>(r, ssig, catt, Kp, hc_bias, x,
                                               out);
}
